// VectorQuantizerEMA_83288005804952
// MI455X (gfx1250) — matrix operand order repaired and checked by reading, not compiled, not run on hardware
//
#include <hip/hip_runtime.h>

typedef __attribute__((ext_vector_type(16))) _Float16 v16h;
typedef __attribute__((ext_vector_type(8)))  float    v8f;

#define NUM_CODES 1024
#define CODE_DIM  256
#define N_ROWS    32768            // 32*512*1*2 rows of 256
#define EMA       0.99f
#define ONE_M_EMA 0.01f
#define COMMIT    0.25f
#define EPSV      1e-6f

// LDS row stride for staged codes: 512B data + 16B pad -> conflict-free B reads
#define LDS_ROW_BYTES 528

// ---------------- ws layout (bytes) ----------------
#define WS_EMBH   0                          // 1024*256 f16      = 524288
#define WS_ENORM  524288                     // 1024 f32          = 4096
#define WS_IDX    528384                     // 32768 i32         = 131072
#define WS_COUNTS 659456                     // 1024 f32          = 4096   (zeroed)
#define WS_ESUM   663552                     // 1024*256 f32      = 1048576(zeroed)
#define WS_LOSS   1712128                    // 1 f32             (zeroed)
#define WS_NSUM   1712132                    // 1 f32             (zeroed)
#define WS_ZERO_OFF  WS_COUNTS
#define WS_ZERO_LEN  (4096 + 1048576 + 16)

// ---------------- out layout (f32 elements) ----------------
#define OUT_ZQ    0            // 8388608
#define OUT_LOSS  8388608      // 1
#define OUT_IDX   8388609      // 32768
#define OUT_NCS   8421377      // 1024
#define OUT_NAVG  8422401      // 262144
#define OUT_ENRM  8684545      // 262144

// CDNA5 async global->LDS staging path (ASYNCcnt), with sync fallback.
#if defined(__has_builtin)
#if __has_builtin(__builtin_amdgcn_global_load_async_to_lds_b128) && \
    __has_builtin(__builtin_amdgcn_s_wait_asynccnt)
#define VQ_ASYNC_LDS 1
#endif
#endif
#ifndef VQ_ASYNC_LDS
#define VQ_ASYNC_LDS 0
#endif

// Match the builtin's parameter types exactly: vector_size(16) int, AS1 / AS3.
typedef int vsi4 __attribute__((__vector_size__(16)));
typedef __attribute__((address_space(1))) vsi4* gv4p;
typedef __attribute__((address_space(3))) vsi4* lv4p;

// Convert codebook to f16 and compute ||e||^2 (exact, from f32).
// grid 128 x 256 threads; one wave per code row.
__global__ void vq_prep_codes(const float* __restrict__ emb,
                              _Float16* __restrict__ embh,
                              float* __restrict__ enorm) {
    int wave = threadIdx.x >> 5, lane = threadIdx.x & 31;
    int code = blockIdx.x * 8 + wave;
    const float* src = emb + (size_t)code * CODE_DIM + lane * 8;
    float4 f0 = *(const float4*)(src);
    float4 f1 = *(const float4*)(src + 4);
    _Float16* dst = embh + (size_t)code * CODE_DIM + lane * 8;
    dst[0]=(_Float16)f0.x; dst[1]=(_Float16)f0.y; dst[2]=(_Float16)f0.z; dst[3]=(_Float16)f0.w;
    dst[4]=(_Float16)f1.x; dst[5]=(_Float16)f1.y; dst[6]=(_Float16)f1.z; dst[7]=(_Float16)f1.w;
    float ss = f0.x*f0.x + f0.y*f0.y + f0.z*f0.z + f0.w*f0.w
             + f1.x*f1.x + f1.y*f1.y + f1.z*f1.z + f1.w*f1.w;
    for (int off = 16; off > 0; off >>= 1) ss += __shfl_xor(ss, off, 32);
    if (lane == 0) enorm[code] = ss;
}

// Stage 64 code rows (32 KB payload) cooperatively into LDS with padded rows.
__device__ __forceinline__ void stage_codes(const _Float16* __restrict__ embh,
                                            char* __restrict__ ldsBase,
                                            int codeBase, int tid) {
    const char* gsrc = (const char*)(embh + (size_t)codeBase * CODE_DIM);
    #pragma unroll
    for (int i = 0; i < 8; ++i) {
        int c    = tid + i * 256;                    // 16B chunk id, 0..2047
        int goff = c * 16;                           // contiguous in global
        int loff = (c >> 5) * LDS_ROW_BYTES + (c & 31) * 16;   // padded in LDS
#if VQ_ASYNC_LDS
        __builtin_amdgcn_global_load_async_to_lds_b128(
            (gv4p)(gsrc + goff), (lv4p)(ldsBase + loff), 0, 0);
#else
        *(uint4*)(ldsBase + loff) = *(const uint4*)(gsrc + goff);
#endif
    }
}

// Nearest-code assignment via f16 WMMA.
// grid 256 x 256 threads; 8 waves/block, each wave owns 16 rows.
// Double-buffered 64-code LDS tiles staged with async global->LDS;
// 4 independent WMMA accumulator chains (one per 16-code tile).
__global__ void vq_assign(const float* __restrict__ z,
                          const _Float16* __restrict__ embh,
                          const float* __restrict__ enorm,
                          int* __restrict__ idxOut) {
    __shared__ __align__(16) char sB[2][64 * LDS_ROW_BYTES];   // 2 x 33 KB

    int tid = threadIdx.x;
    int wave = tid >> 5, lane = tid & 31;
    int g  = lane >> 4;        // half-wave group
    int ml = lane & 15;        // A: row, B/C: column
    int rowBase = (blockIdx.x * 8 + wave) * 16;

    // ---- load A tile: 16 rows x 256 dims, f32 -> f16, WMMA A layout ----
    // lane holds row=ml; per 32-K chunk: halfs 0-15 = K[g*16..+16)
    v16h a[8];
    {
        const float* zrow = z + (size_t)(rowBase + ml) * CODE_DIM;
        for (int kc = 0; kc < 8; ++kc) {
            const float* p1 = zrow + kc * 32 + g * 16;
            const float* p2 = p1 + 8;
            float4 c1a = *(const float4*)p1;
            float4 c1b = *(const float4*)(p1 + 4);
            float4 c2a = *(const float4*)p2;
            float4 c2b = *(const float4*)(p2 + 4);
            v16h av;
            av[0]=(_Float16)c1a.x; av[1]=(_Float16)c1a.y; av[2]=(_Float16)c1a.z; av[3]=(_Float16)c1a.w;
            av[4]=(_Float16)c1b.x; av[5]=(_Float16)c1b.y; av[6]=(_Float16)c1b.z; av[7]=(_Float16)c1b.w;
            av[8]=(_Float16)c2a.x; av[9]=(_Float16)c2a.y; av[10]=(_Float16)c2a.z; av[11]=(_Float16)c2a.w;
            av[12]=(_Float16)c2b.x; av[13]=(_Float16)c2b.y; av[14]=(_Float16)c2b.z; av[15]=(_Float16)c2b.w;
            a[kc] = av;
        }
    }

    float best[8];
    int   bidx[8];
    for (int r = 0; r < 8; ++r) { best[r] = __builtin_inff(); bidx[r] = 0; }

    stage_codes(embh, sB[0], 0, tid);                 // prologue: chunk 0 -> buf 0

    for (int chunk = 0; chunk < 16; ++chunk) {
        int buf = chunk & 1;
        int codeBase = chunk * 64;
#if VQ_ASYNC_LDS
        __builtin_amdgcn_s_wait_asynccnt(0);          // my async loads landed in LDS
#endif
        __syncthreads();                              // everyone's loads + prior reads done
        if (chunk + 1 < 16)
            stage_codes(embh, sB[buf ^ 1], codeBase + 64, tid);   // overlap next fetch

        float en0 = enorm[codeBase + ml];
        float en1 = enorm[codeBase + 16 + ml];
        float en2 = enorm[codeBase + 32 + ml];
        float en3 = enorm[codeBase + 48 + ml];

        // 4 independent accumulator chains over the 4 16-code tiles
        const char* bb = sB[buf] + (size_t)ml * LDS_ROW_BYTES + g * 32;
        v8f c0 = {}, c1 = {}, c2 = {}, c3 = {};
        #pragma unroll
        for (int kc = 0; kc < 8; ++kc) {
            int kOff = kc * 64;                       // bytes within row
            v16h b0 = *(const v16h*)(bb + 0 * 16 * LDS_ROW_BYTES + kOff);
            v16h b1 = *(const v16h*)(bb + 1 * 16 * LDS_ROW_BYTES + kOff);
            v16h b2 = *(const v16h*)(bb + 2 * 16 * LDS_ROW_BYTES + kOff);
            v16h b3 = *(const v16h*)(bb + 3 * 16 * LDS_ROW_BYTES + kOff);
            c0 = __builtin_amdgcn_wmma_f32_16x16x32_f16(false, a[kc], false, b0, (short)0, c0, false, false);
            c1 = __builtin_amdgcn_wmma_f32_16x16x32_f16(false, a[kc], false, b1, (short)0, c1, false, false);
            c2 = __builtin_amdgcn_wmma_f32_16x16x32_f16(false, a[kc], false, b2, (short)0, c2, false, false);
            c3 = __builtin_amdgcn_wmma_f32_16x16x32_f16(false, a[kc], false, b3, (short)0, c3, false, false);
        }

        #pragma unroll
        for (int r = 0; r < 8; ++r) {
            float s0 = en0 - 2.0f * c0[r];            // ||e||^2 - 2 x.e (argmin-equivalent)
            float s1 = en1 - 2.0f * c1[r];
            float s2 = en2 - 2.0f * c2[r];
            float s3 = en3 - 2.0f * c3[r];
            if (s0 < best[r]) { best[r] = s0; bidx[r] = codeBase + ml; }
            if (s1 < best[r]) { best[r] = s1; bidx[r] = codeBase + 16 + ml; }
            if (s2 < best[r]) { best[r] = s2; bidx[r] = codeBase + 32 + ml; }
            if (s3 < best[r]) { best[r] = s3; bidx[r] = codeBase + 48 + ml; }
        }
    }

    // argmin across the 16 lanes of each column group (xor 1/2/4/8 stays in group)
    for (int r = 0; r < 8; ++r) {
        float bs = best[r]; int bi = bidx[r];
        for (int off = 8; off > 0; off >>= 1) {
            float os = __shfl_xor(bs, off, 32);
            int   oi = __shfl_xor(bi, off, 32);
            if (os < bs || (os == bs && oi < bi)) { bs = os; bi = oi; }
        }
        if (ml == 0) idxOut[rowBase + g * 8 + r] = bi;   // row = rowBase + 8g + r
    }
}

// Gather z_q (exact f32), loss, histogram + scatter sums, idx as float.
// grid 4096 x 256; one wave per row.
__global__ void vq_gather_scatter(const float* __restrict__ z,
                                  const float* __restrict__ emb,
                                  const int* __restrict__ idx,
                                  float* __restrict__ zqOut,
                                  float* __restrict__ idxOut,
                                  float* __restrict__ counts,
                                  float* __restrict__ esum,
                                  float* __restrict__ lossAcc) {
    int wave = threadIdx.x >> 5, lane = threadIdx.x & 31;
    int row = blockIdx.x * 8 + wave;
    int k = idx[row];
    const float* zr = z   + (size_t)row * CODE_DIM + lane * 8;
    const float* er = emb + (size_t)k   * CODE_DIM + lane * 8;
    float4 z0 = *(const float4*)zr,       z1 = *(const float4*)(zr + 4);
    float4 e0 = *(const float4*)er,       e1 = *(const float4*)(er + 4);
    float* zq = zqOut + (size_t)row * CODE_DIM + lane * 8;
    *(float4*)zq       = e0;
    *(float4*)(zq + 4) = e1;

    float d0=e0.x-z0.x, d1=e0.y-z0.y, d2=e0.z-z0.z, d3=e0.w-z0.w;
    float d4=e1.x-z1.x, d5=e1.y-z1.y, d6=e1.z-z1.z, d7=e1.w-z1.w;
    float l = d0*d0+d1*d1+d2*d2+d3*d3+d4*d4+d5*d5+d6*d6+d7*d7;
    for (int off = 16; off > 0; off >>= 1) l += __shfl_xor(l, off, 32);

    float* es = esum + (size_t)k * CODE_DIM + lane * 8;
    atomicAdd(es + 0, z0.x); atomicAdd(es + 1, z0.y);
    atomicAdd(es + 2, z0.z); atomicAdd(es + 3, z0.w);
    atomicAdd(es + 4, z1.x); atomicAdd(es + 5, z1.y);
    atomicAdd(es + 6, z1.z); atomicAdd(es + 7, z1.w);

    if (lane == 0) {
        atomicAdd(lossAcc, l);
        atomicAdd(&counts[k], 1.0f);
        idxOut[row] = (float)k;
    }
}

// EMA cluster sizes + total n + loss finalize. Single block of 1024.
__global__ void vq_ema_cs(const float* __restrict__ csIn,
                          const float* __restrict__ counts,
                          float* __restrict__ ncsOut,
                          float* __restrict__ nSum,
                          const float* __restrict__ lossAcc,
                          float* __restrict__ lossOut) {
    __shared__ float red[32];
    int t = threadIdx.x, lane = t & 31, wave = t >> 5;
    float ncs = EMA * csIn[t] + ONE_M_EMA * counts[t];
    ncsOut[t] = ncs;
    float s = ncs;
    for (int off = 16; off > 0; off >>= 1) s += __shfl_xor(s, off, 32);
    if (lane == 0) red[wave] = s;
    __syncthreads();
    if (t < 32) {
        float v = red[t];
        for (int off = 16; off > 0; off >>= 1) v += __shfl_xor(v, off, 32);
        if (t == 0) {
            nSum[0] = v;
            lossOut[0] = lossAcc[0] * (COMMIT / (float)(N_ROWS * CODE_DIM));
        }
    }
}

// EMA embedding_avg + dead-code normalization. grid 1024 x 256.
__global__ void vq_ema_avg(const float* __restrict__ avgIn,
                           const float* __restrict__ esum,
                           const float* __restrict__ ncs,
                           const float* __restrict__ nSum,
                           float* __restrict__ navgOut,
                           float* __restrict__ enrmOut) {
    int k = blockIdx.x, d = threadIdx.x;
    size_t i = (size_t)k * CODE_DIM + d;
    float na = EMA * avgIn[i] + ONE_M_EMA * esum[i];
    navgOut[i] = na;
    float n = nSum[0];
    float cs = (ncs[k] + EPSV) / (n + NUM_CODES * EPSV) * n;
    cs = fmaxf(cs, EPSV);
    enrmOut[i] = na / cs;
}

extern "C" void kernel_launch(void* const* d_in, const int* in_sizes, int n_in,
                              void* d_out, int out_size, void* d_ws, size_t ws_size,
                              hipStream_t stream) {
    const float* z    = (const float*)d_in[0];
    const float* emb  = (const float*)d_in[1];
    const float* csIn = (const float*)d_in[2];
    const float* avgI = (const float*)d_in[3];

    char* ws = (char*)d_ws;
    _Float16* embh  = (_Float16*)(ws + WS_EMBH);
    float* enorm    = (float*)(ws + WS_ENORM);
    int*   idx      = (int*)  (ws + WS_IDX);
    float* counts   = (float*)(ws + WS_COUNTS);
    float* esum     = (float*)(ws + WS_ESUM);
    float* lossAcc  = (float*)(ws + WS_LOSS);
    float* nSum     = (float*)(ws + WS_NSUM);

    float* out = (float*)d_out;
    float* zqOut   = out + OUT_ZQ;
    float* lossOut = out + OUT_LOSS;
    float* idxOut  = out + OUT_IDX;
    float* ncsOut  = out + OUT_NCS;
    float* navgOut = out + OUT_NAVG;
    float* enrmOut = out + OUT_ENRM;

    (void)hipMemsetAsync(ws + WS_ZERO_OFF, 0, WS_ZERO_LEN, stream);

    vq_prep_codes<<<128, 256, 0, stream>>>(emb, embh, enorm);
    vq_assign<<<256, 256, 0, stream>>>(z, embh, enorm, idx);
    vq_gather_scatter<<<4096, 256, 0, stream>>>(z, emb, idx, zqOut, idxOut,
                                                counts, esum, lossAcc);
    vq_ema_cs<<<1, 1024, 0, stream>>>(csIn, counts, ncsOut, nSum, lossAcc, lossOut);
    vq_ema_avg<<<NUM_CODES, CODE_DIM, 0, stream>>>(avgI, esum, ncsOut, nSum,
                                                   navgOut, enrmOut);
}
